// SegMetric_37538014167087
// MI455X (gfx1250) — 8-bit matrix operand order repaired and checked by reading, compiled, not run on hardware
//
#include <hip/hip_runtime.h>
#include <hip/hip_bf16.h>
#include <stdint.h>

#define NUM_CLASSES 19
#define HW 262144
#define B_SZ 16
#define CONF_DIM 32
#define CONF_SZ (CONF_DIM * CONF_DIM)
#define SEGS_PER_SAMPLE 64
#define THREADS 128
#define WAVES (THREADS / 32)
#define PIX_PER_WG (HW / SEGS_PER_SAMPLE)
#define PIX_PER_WAVE (PIX_PER_WG / WAVES)
#define CHUNK 128
#define CHUNKS_PER_WAVE (PIX_PER_WAVE / CHUNK)
#define OH_BYTES 8192
#define FP8_ONE 0x38

typedef int   v16i __attribute__((ext_vector_type(16)));
typedef float v8f  __attribute__((ext_vector_type(8)));

__global__ void SegMetric_zero_kernel(unsigned* __restrict__ p, int n) {
  int i = blockIdx.x * blockDim.x + threadIdx.x;
  if (i < n) p[i] = 0u;
}

__device__ __forceinline__ v16i load_frag(const unsigned char* p) {
  const uint4* f = (const uint4*)p;
  uint4 q0 = f[0], q1 = f[1], q2 = f[2], q3 = f[3];
  v16i r;
  r[0]  = (int)q0.x; r[1]  = (int)q0.y; r[2]  = (int)q0.z; r[3]  = (int)q0.w;
  r[4]  = (int)q1.x; r[5]  = (int)q1.y; r[6]  = (int)q1.z; r[7]  = (int)q1.w;
  r[8]  = (int)q2.x; r[9]  = (int)q2.y; r[10] = (int)q2.z; r[11] = (int)q2.w;
  r[12] = (int)q3.x; r[13] = (int)q3.y; r[14] = (int)q3.z; r[15] = (int)q3.w;
  return r;
}

__global__ __launch_bounds__(THREADS) void SegMetric_conf_kernel(
    const float* __restrict__ X, const float* __restrict__ S,
    unsigned* __restrict__ gconf) {
  __shared__ unsigned conf[CONF_SZ];
  __shared__ __align__(64) unsigned char oh[WAVES][OH_BYTES];

  const int tid  = threadIdx.x;
  const int wave = tid >> 5;
  const int lane = tid & 31;
  const int h    = lane >> 4;
  const int nl   = lane & 15;

  const int wg  = blockIdx.x;
  const int b   = wg / SEGS_PER_SAMPLE;
  const int seg = wg % SEGS_PER_SAMPLE;

#pragma unroll
  for (int k = 0; k < CONF_SZ / THREADS; ++k) conf[tid + k * THREADS] = 0u;
  __syncthreads();

  const size_t sampleOff = (size_t)b * NUM_CLASSES * HW;
  const float* xb = X + sampleOff;
  const float* sb = S + sampleOff;

  unsigned char* mybase = &oh[wave][0];

  const int hA = (lane >> 1) & 1;
  const int vA = (lane & 1) | (((lane >> 2) & 1) << 1) | (((lane >> 3) & 1) << 2) | ((lane >> 4) << 3);
  const int constA = hA * 1024 + vA * 4;
  const int hB = (lane >> 1) & 1;
  const int vB = (lane & 1) | (((lane >> 2) & 1) << 1) | (((lane >> 3) & 1) << 2) | ((lane >> 4) << 3);
  const int constB = 4096 + hB * 1024 + vB * 4;

  v8f d00 = {}; v8f d01 = {}; v8f d10 = {}; v8f d11 = {};

  const int waveBase = seg * PIX_PER_WG + wave * PIX_PER_WAVE;

  for (int ch = 0; ch < CHUNKS_PER_WAVE; ++ch) {
    const int p0 = waveBase + ch * CHUNK + 4 * lane;

    float4 vx = *(const float4*)(xb + p0);
    float4 vs = *(const float4*)(sb + p0);
    float bx0 = vx.x, bx1 = vx.y, bx2 = vx.z, bx3 = vx.w;
    float bs0 = vs.x, bs1 = vs.y, bs2 = vs.z, bs3 = vs.w;
    int px[4] = {0, 0, 0, 0};
    int pg[4] = {0, 0, 0, 0};
#pragma unroll
    for (int c = 1; c < NUM_CLASSES; ++c) {
      const float4 a = *(const float4*)(xb + (size_t)c * HW + p0);
      if (a.x > bx0) { bx0 = a.x; px[0] = c; }
      if (a.y > bx1) { bx1 = a.y; px[1] = c; }
      if (a.z > bx2) { bx2 = a.z; px[2] = c; }
      if (a.w > bx3) { bx3 = a.w; px[3] = c; }
      const float4 g = *(const float4*)(sb + (size_t)c * HW + p0);
      if (g.x > bs0) { bs0 = g.x; pg[0] = c; }
      if (g.y > bs1) { bs1 = g.y; pg[1] = c; }
      if (g.z > bs2) { bs2 = g.z; pg[2] = c; }
      if (g.w > bs3) { bs3 = g.w; pg[3] = c; }
    }

    const uint4 z = make_uint4(0u, 0u, 0u, 0u);
#pragma unroll
    for (int i = 0; i < OH_BYTES / 32 / 16; ++i)
      *(uint4*)(mybase + lane * (OH_BYTES / 32) + i * 16) = z;
    asm volatile("" ::: "memory");

#pragma unroll
    for (int j = 0; j < 4; ++j) {
      const int g = pg[j];
      const int p = px[j];
      mybase[(g << 6) + ((g >> 4) << 10) + constA + j] = FP8_ONE;
      mybase[(p << 6) + ((p >> 4) << 10) + constB + j] = FP8_ONE;
    }
    asm volatile("" ::: "memory");

    const v16i a0 = load_frag(mybase +        lane * 64);
    const v16i a1 = load_frag(mybase + 2048 + lane * 64);
    const v16i b0 = load_frag(mybase + 4096 + lane * 64);
    const v16i b1 = load_frag(mybase + 6144 + lane * 64);

    d00 = __builtin_amdgcn_wmma_f32_16x16x128_fp8_fp8(a0, b0, (short)0, d00, false, false);
    d01 = __builtin_amdgcn_wmma_f32_16x16x128_fp8_fp8(a0, b1, (short)0, d01, false, false);
    d10 = __builtin_amdgcn_wmma_f32_16x16x128_fp8_fp8(a1, b0, (short)0, d10, false, false);
    d11 = __builtin_amdgcn_wmma_f32_16x16x128_fp8_fp8(a1, b1, (short)0, d11, false, false);
    asm volatile("" ::: "memory");
  }

#pragma unroll
  for (int r = 0; r < 8; ++r) {
    const int m0 = r + 8 * h;
    atomicAdd(&conf[(m0     ) * CONF_DIM + nl     ], (unsigned)d00[r]);
    atomicAdd(&conf[(m0     ) * CONF_DIM + nl + 16], (unsigned)d01[r]);
    atomicAdd(&conf[(m0 + 16) * CONF_DIM + nl     ], (unsigned)d10[r]);
    atomicAdd(&conf[(m0 + 16) * CONF_DIM + nl + 16], (unsigned)d11[r]);
  }
  __syncthreads();

  unsigned* gc = gconf + (size_t)b * CONF_SZ;
#pragma unroll
  for (int k = 0; k < CONF_SZ / THREADS; ++k) {
    const unsigned v = conf[tid + k * THREADS];
    if (v) atomicAdd(gc + tid + k * THREADS, v);
  }
}

__global__ __launch_bounds__(512) void SegMetric_finalize_kernel(
    const unsigned* __restrict__ gconf, float* __restrict__ out) {
  __shared__ float siou[B_SZ * NUM_CLASSES];
  const int t = threadIdx.x;
  if (t < B_SZ * NUM_CLASSES) {
    const int b = t / NUM_CLASSES;
    const int c = t % NUM_CLASSES;
    const unsigned* g = gconf + (size_t)b * CONF_SZ;
    const float diag = (float)g[c * CONF_DIM + c];
    float row = 0.f, col = 0.f;
#pragma unroll
    for (int j = 0; j < NUM_CLASSES; ++j) {
      row += (float)g[c * CONF_DIM + j];
      col += (float)g[j * CONF_DIM + c];
    }
    siou[t] = diag / (row + col - diag + 1e-10f);
  }
  __syncthreads();
  if (t == 0) {
    float acc = 0.f;
    for (int b = 0; b < B_SZ; ++b) {
      float s = 0.f;
      for (int c = 0; c < NUM_CLASSES; ++c) s += siou[b * NUM_CLASSES + c];
      acc += s / (float)NUM_CLASSES;
    }
    out[0] = acc / (float)B_SZ;
  }
}

extern "C" void kernel_launch(void* const* d_in, const int* in_sizes, int n_in,
                              void* d_out, int out_size, void* d_ws, size_t ws_size,
                              hipStream_t stream) {
  (void)in_sizes; (void)n_in; (void)out_size; (void)ws_size;
  const float* X = (const float*)d_in[0];
  const float* S = (const float*)d_in[1];
  unsigned* gconf = (unsigned*)d_ws;
  float* out = (float*)d_out;

  const int n = B_SZ * CONF_SZ;
  SegMetric_zero_kernel<<<(n + 255) / 256, 256, 0, stream>>>(gconf, n);
  SegMetric_conf_kernel<<<B_SZ * SEGS_PER_SAMPLE, THREADS, 0, stream>>>(X, S, gconf);
  SegMetric_finalize_kernel<<<1, 512, 0, stream>>>(gconf, out);
}
